// StagedVQVAE_67302137528708
// MI455X (gfx1250) — matrix operand order repaired and checked by reading, not compiled, not run on hardware
//
#include <hip/hip_runtime.h>
#include <hip/hip_bf16.h>
#include <math.h>

typedef __attribute__((ext_vector_type(16))) _Float16 v16h;
typedef __attribute__((ext_vector_type(8)))  _Float16 v8h;
typedef __attribute__((ext_vector_type(8)))  float    v8f;

#define ACT_NONE  0
#define ACT_SWISH 1

__device__ __forceinline__ float swish_f(float y) {
  return y / (1.0f + __expf(-y));
}

__device__ __forceinline__ v8f wmma16x16x32(v16h a, v16h b, v8f c) {
  return __builtin_amdgcn_wmma_f32_16x16x32_f16(
      false, a, false, b, (short)0, c, false, false);
}

// ===========================================================================
// Fragment-order layouts (every fragment is one contiguous 32B object):
//   B-fragment element i of lane L  <->  K = kc*32 + (L>>4)*16 + i,
//                                         N = ntile*16 + (L&15)
//   A-fragment element i of lane L  <->  M = L&15,
//                                         K = kc*32 + (L>>4)*16 + i
// Packed weights / codebook: [tile][kchunk][lane][16] halves.
// ===========================================================================

// ---------------------------------------------------------------------------
// Implicit-GEMM causal conv / phase-decomposed conv-transpose, f16 WMMA.
//  MODE 0: causal conv, out[t,n] = sum_{tap,c} in[t*stride-(KSIZE-1)+tap, c]*w
//  MODE 1: convT phase: rows are q; out t = q*stride+phase; K=2*Cin.
//  Block = 128 threads (4 waves); block tile 128(M) x 16(N); each wave owns
//  two 16-row subtiles so one B fragment feeds two WMMAs (register blocking).
//  A staged per-wave in LDS in fragment order (16B vector runs); B fragments
//  streamed directly from L2-resident fragment-packed weights (2x b128/WMMA
//  pair).  No barriers: no cross-wave LDS sharing; same-wave DS is in-order.
//  grid = (rows/128, Cout/16, B)
// ---------------------------------------------------------------------------
template <int MODE, int CG, int KSIZE, bool IN_F16, bool OUT_F16>
__global__ void __launch_bounds__(128) conv_wmma_kernel(
    const void* __restrict__ in_, const _Float16* __restrict__ wfrag,
    const float* __restrict__ bias, const _Float16* __restrict__ resid,
    void* __restrict__ out_,
    int Ti, int To, int Cin, int Cout, int stride, int phase, int act)
{
  constexpr int Kt = (MODE == 0) ? (CG * KSIZE) : (2 * CG);
  constexpr int KC = (Kt + 31) / 32;

  __shared__ __align__(32) _Float16 As[4][2][32 * 16];  // [wave][subtile]

  const int tid  = threadIdx.x;
  const int wave = tid >> 5;
  const int lane = tid & 31;
  const int hl   = lane >> 4;
  const int lrow = lane & 15;

  const int m0 = blockIdx.x * 128 + wave * 32;   // two subtiles: m0, m0+16
  const int n0 = blockIdx.y * 16;
  const int b  = blockIdx.z;

  const int G  = Cin / CG;                 // feature groups (1 for dense)
  const int Og = Cout / G;
  const int g  = n0 / Og;
  const int NT = Og >> 4;
  const int nt = (n0 - g * Og) >> 4;

  const float*    inF = (const float*)in_;
  const _Float16* inH = (const _Float16*)in_;

  // base of this tile's fragment-packed weights; +512 halves per k-chunk
  const _Float16* wp = wfrag + ((((size_t)g * NT + nt) * KC) * 32 + lane) * 16;

  v8f acc0 = {0.f, 0.f, 0.f, 0.f, 0.f, 0.f, 0.f, 0.f};
  v8f acc1 = acc0;

  for (int kc = 0; kc < KC; ++kc) {
    const int kk = kc * 32;
    // --- stage both A subtiles in fragment order: 2 runs/lane/subtile ---
    #pragma unroll
    for (int sub = 0; sub < 2; ++sub) {
      #pragma unroll
      for (int rr = 0; rr < 2; ++rr) {
        const int r   = rr * 32 + lane;     // run id 0..63
        const int l   = r >> 1;             // dest lane slot
        const int i0  = (r & 1) * 8;        // dest element base
        const int M   = (l & 15) + sub * 16;
        const int hlr = l >> 4;
        const int Kb  = kk + hlr * 16 + i0;             // run start K
        v8h hv;
        #pragma unroll
        for (int q = 0; q < 8; ++q) hv[q] = (_Float16)0.f;
        if (Kb < Kt) {                      // Kt % 8 == 0 -> all-or-none run
          int ti, c0;
          if (MODE == 0) {
            const int tap = Kb / CG;        // constexpr divisor
            c0 = g * CG + (Kb - tap * CG);
            ti = (m0 + M) * stride - (KSIZE - 1) + tap;
          } else {
            const int ts = Kb / CG;
            c0 = Kb - ts * CG;
            ti = (m0 + M) - ts;
          }
          if (ti >= 0 && ti < Ti) {         // run stays inside one tap (CG%8==0)
            const size_t ai = ((size_t)b * Ti + ti) * Cin + c0;
            if constexpr (IN_F16) {
              hv = *(const v8h*)(inH + ai);               // global_load_b128
            } else {
              const float4 f0 = *(const float4*)(inF + ai);
              const float4 f1 = *(const float4*)(inF + ai + 4);
              hv[0] = (_Float16)f0.x; hv[1] = (_Float16)f0.y;
              hv[2] = (_Float16)f0.z; hv[3] = (_Float16)f0.w;
              hv[4] = (_Float16)f1.x; hv[5] = (_Float16)f1.y;
              hv[6] = (_Float16)f1.z; hv[7] = (_Float16)f1.w;
            }
          }
        }
        *(v8h*)(&As[wave][sub][l * 16 + i0]) = hv;        // ds_store_b128
      }
    }
    // same-wave DS is in-order; no barrier needed (no cross-wave sharing)
    const v16h af0 = *(const v16h*)(&As[wave][0][lane * 16]);  // ds_load_b128
    const v16h af1 = *(const v16h*)(&As[wave][1][lane * 16]);
    const v16h bf  = *(const v16h*)(wp + (size_t)kc * 512);    // global b128
    acc0 = wmma16x16x32(af0, bf, acc0);      // one B fragment, two WMMAs
    acc1 = wmma16x16x32(af1, bf, acc1);
  }

  // --- fused epilogue: bias, residual, swish ---
  const int n = n0 + lrow;
  const float bn = bias[n];
  #pragma unroll
  for (int sub = 0; sub < 2; ++sub) {
    const v8f acc = sub ? acc1 : acc0;
    #pragma unroll
    for (int v = 0; v < 8; ++v) {
      const int m = m0 + sub * 16 + v + hl * 8;     // C/D layout row
      const int t = (MODE == 0) ? m : (m * stride + phase);
      if (t < To) {
        const size_t oi = ((size_t)b * To + t) * Cout + n;
        float y = acc[v] + bn;
        if (resid) y += (float)resid[oi];
        if (act == ACT_SWISH) y = swish_f(y);
        if constexpr (OUT_F16) ((_Float16*)out_)[oi] = (_Float16)y;
        else                   ((float*)out_)[oi]    = y;
      }
    }
  }
}

// ---------------------------------------------------------------------------
// VQ nearest-neighbor: per group g, score = |c|^2 - 2 x.c via WMMA GEMM.
// Codebook fragment-packed [g][cc][kc][lane][16] f16; x f32 rows.
// One wave = 16 rows x 1024 codes; A fragments hoisted across all chunks.
// grid = (rows/16, G)
// ---------------------------------------------------------------------------
__global__ void __launch_bounds__(32) vq_argmin_kernel(
    const float* __restrict__ x, const _Float16* __restrict__ cbfrag,
    const float* __restrict__ cnorm, int* __restrict__ idx_out,
    float* __restrict__ idx_out_f, int rows, int D)
{
  constexpr int CD = 128, CS = 1024, NK = CD / 32, NCC = CS / 16;

  const int lane = threadIdx.x;
  const int hl   = lane >> 4;
  const int lrow = lane & 15;
  const int r0 = blockIdx.x * 16;
  const int g  = blockIdx.y;
  const int G  = gridDim.y;

  // A fragments: 2 vectorized f32 runs of 8 per k-chunk, converted to f16
  v16h af[NK];
  const float* xb = x + ((size_t)(r0 + lrow)) * D + g * CD;
  #pragma unroll
  for (int kc = 0; kc < NK; ++kc) {
    #pragma unroll
    for (int rr = 0; rr < 2; ++rr) {
      const int Kb = kc * 32 + hl * 16 + rr * 8;
      const float4 f0 = *(const float4*)(xb + Kb);
      const float4 f1 = *(const float4*)(xb + Kb + 4);
      af[kc][rr * 8 + 0] = (_Float16)f0.x; af[kc][rr * 8 + 1] = (_Float16)f0.y;
      af[kc][rr * 8 + 2] = (_Float16)f0.z; af[kc][rr * 8 + 3] = (_Float16)f0.w;
      af[kc][rr * 8 + 4] = (_Float16)f1.x; af[kc][rr * 8 + 5] = (_Float16)f1.y;
      af[kc][rr * 8 + 6] = (_Float16)f1.z; af[kc][rr * 8 + 7] = (_Float16)f1.w;
    }
  }

  float best[8];
  int   bidx[8];
  #pragma unroll
  for (int v = 0; v < 8; ++v) { best[v] = 3.4e38f; bidx[v] = 0; }

  const _Float16* cb = cbfrag + (((size_t)g * NCC * NK) * 32 + lane) * 16;

  for (int cc = 0; cc < NCC; ++cc) {
    v8f acc = {0.f, 0.f, 0.f, 0.f, 0.f, 0.f, 0.f, 0.f};
    #pragma unroll
    for (int kc = 0; kc < NK; ++kc) {
      const v16h bf = *(const v16h*)(cb + ((size_t)cc * NK + kc) * 512);
      acc = wmma16x16x32(af[kc], bf, acc);
    }
    const int code = cc * 16 + lrow;
    const float cn = cnorm[g * CS + code];
    #pragma unroll
    for (int v = 0; v < 8; ++v) {
      const float s = cn - 2.0f * acc[v];     // |x|^2 constant per row
      if (s < best[v]) { best[v] = s; bidx[v] = code; }
    }
  }

  __shared__ float sv[16][16];
  __shared__ int   si[16][16];
  #pragma unroll
  for (int v = 0; v < 8; ++v) {
    const int M = v + hl * 8;
    sv[M][lrow] = best[v];
    si[M][lrow] = bidx[v];
  }
  __syncthreads();
  if (lane < 16) {
    float bv = sv[lane][0];
    int   bi = si[lane][0];
    for (int j = 1; j < 16; ++j)
      if (sv[lane][j] < bv) { bv = sv[lane][j]; bi = si[lane][j]; }
    const int r = r0 + lane;
    idx_out  [(size_t)r * G + g] = bi;
    idx_out_f[(size_t)r * G + g] = (float)bi;
  }
}

// --------------------- fragment-order pack kernels -------------------------

// conv weights [O, Cg, k] f32 -> [g][nt][kc][lane][16] f16
__global__ void pack_convB_kernel(const float* __restrict__ w,
                                  _Float16* __restrict__ o,
                                  int Cout, int Cg, int ks, int G, int n) {
  int e = blockIdx.x * blockDim.x + threadIdx.x;
  if (e >= n) return;
  const int Og = Cout / G, Kt = Cg * ks, KC = (Kt + 31) / 32;
  const int i    = e & 15;
  const int lane = (e >> 4) & 31;
  int rest = e >> 9;
  const int kc = rest % KC; rest /= KC;
  const int NT = Og >> 4;
  const int nt = rest % NT; rest /= NT;
  const int g  = rest;
  const int K  = kc * 32 + (lane >> 4) * 16 + i;
  _Float16 val = (_Float16)0.f;
  if (K < Kt) {
    const int tap = K / Cg, cg = K - tap * Cg;
    const int oo = g * Og + nt * 16 + (lane & 15);
    val = (_Float16)w[((size_t)oo * Cg + cg) * ks + tap];
  }
  o[e] = val;
}

// convT weights [I, O, 2s] f32 -> [p][nt][kc][lane][16] f16 (K = ts*Cin + c)
__global__ void pack_convTB_kernel(const float* __restrict__ w,
                                   _Float16* __restrict__ o,
                                   int Cin, int Cout, int s, int n) {
  int e = blockIdx.x * blockDim.x + threadIdx.x;
  if (e >= n) return;
  const int Kt = 2 * Cin, KC = Kt / 32, NT = Cout >> 4;
  const int i    = e & 15;
  const int lane = (e >> 4) & 31;
  int rest = e >> 9;
  const int kc = rest % KC; rest /= KC;
  const int nt = rest % NT; rest /= NT;
  const int p  = rest;
  const int K  = kc * 32 + (lane >> 4) * 16 + i;
  const int ts = K / Cin, c = K - ts * Cin;
  const int oo = nt * 16 + (lane & 15);
  const int j  = p + ts * s;
  o[e] = (_Float16)w[((size_t)c * Cout + oo) * (2 * s) + j];
}

// codebook [G, cs, cd] f32 -> [g][cc][kc][lane][16] f16
__global__ void pack_cb_kernel(const float* __restrict__ cb,
                               _Float16* __restrict__ o, int n) {
  int e = blockIdx.x * blockDim.x + threadIdx.x;
  if (e >= n) return;
  constexpr int NK = 4, NCC = 64, CS = 1024, CD = 128;
  const int i    = e & 15;
  const int lane = (e >> 4) & 31;
  int rest = e >> 9;
  const int kc = rest % NK; rest /= NK;
  const int cc = rest % NCC; rest /= NCC;
  const int g  = rest;
  const int code = cc * 16 + (lane & 15);
  const int k    = kc * 32 + (lane >> 4) * 16 + i;
  o[e] = (_Float16)cb[((size_t)g * CS + code) * CD + k];
}

__global__ void cnorm_kernel(const float* __restrict__ cb,
                             float* __restrict__ cn, int cd, int total) {
  int i = blockIdx.x * blockDim.x + threadIdx.x;
  if (i >= total) return;
  const float* p = cb + (size_t)i * cd;
  float s = 0.f;
  for (int k = 0; k < cd; ++k) { float w = p[k]; s += w * w; }
  cn[i] = s;
}

__global__ void avgpool_kernel(const float* __restrict__ in,
                               float* __restrict__ out,
                               int Tin, int D, int rate, int total) {
  int i = blockIdx.x * blockDim.x + threadIdx.x;
  if (i >= total) return;
  const int Tq = Tin / rate;
  const int b = i / (Tq * D), rem = i % (Tq * D);
  const int tq = rem / D, d = rem % D;
  const float* p = in + ((size_t)b * Tin + (size_t)tq * rate) * D + d;
  float s = 0.f;
  for (int k = 0; k < rate; ++k) s += p[(size_t)k * D];
  out[i] = s / (float)rate;
}

__global__ void __launch_bounds__(256) vq_loss_kernel(
    const float* __restrict__ xd, const float* __restrict__ cb,
    const int* __restrict__ idx, float* __restrict__ acc,
    int rows, int G, int cs, int cd) {
  const int total = rows * G * cd;
  int i = blockIdx.x * 256 + threadIdx.x;
  float d = 0.f;
  if (i < total) {
    const int cgd = G * cd;
    const int r = i / cgd, rem = i % cgd;
    const int g = rem / cd, k = rem % cd;
    const int id = idx[(size_t)r * G + g];
    const float q = cb[((size_t)g * cs + id) * cd + k];
    const float df = q - xd[i];
    d = df * df;
  }
  __shared__ float sbuf[256];
  sbuf[threadIdx.x] = d;
  __syncthreads();
  for (int s = 128; s > 0; s >>= 1) {
    if (threadIdx.x < s) sbuf[threadIdx.x] += sbuf[threadIdx.x + s];
    __syncthreads();
  }
  if (threadIdx.x == 0) atomicAdd(acc, sbuf[0]);
}

__global__ void vq_update_kernel(float* __restrict__ residual,
                                 float* __restrict__ quant,
                                 const float* __restrict__ cb,
                                 const int* __restrict__ idx,
                                 int T, int D, int rate, int G, int cs, int cd,
                                 int total) {
  int i = blockIdx.x * blockDim.x + threadIdx.x;
  if (i >= total) return;
  const int b = i / (T * D), rem = i % (T * D);
  const int t = rem / D, d = rem % D;
  const int g = d / cd;
  const int Tq = T / rate;
  const int id = idx[((size_t)b * Tq + t / rate) * G + g];
  const float q = cb[((size_t)g * cs + id) * cd + (d - g * cd)];
  residual[i] -= q;
  quant[i] += q;
}

__global__ void finalize_loss_kernel(const float* __restrict__ lacc,
                                     float* __restrict__ out,
                                     float n0, float n1) {
  if (blockIdx.x == 0 && threadIdx.x == 0)
    out[0] = 0.5f * (lacc[0] / n0 + lacc[1] / n1);
}

// ---------------------------------------------------------------------------

extern "C" void kernel_launch(void* const* d_in, const int* in_sizes, int n_in,
                              void* d_out, int out_size, void* d_ws, size_t ws_size,
                              hipStream_t stream) {
  (void)in_sizes; (void)n_in; (void)out_size; (void)ws_size;
  const int B = 8, T = 4096, C = 80, D = 512, GR = 32;

  const float* x = (const float*)d_in[0];
  // params leaves flattened in sorted-dict order: dec, enc, vq.
  // dec.b0.r1.{c1_b,c1_w,c2_b,c2_w}=0..3, dec.b0.r2=4..7, up_b=8, up_w=9,
  // dec.b1.*=10..19, dec.in_b=20, in_w=21, out_b=22, out_w=23,
  // enc.b0.down_b=24, down_w=25, r1=26..29, r2=30..33,
  // enc.b1.down_b=34, down_w=35, r1=36..39, r2=40..43,
  // enc.in_b=44, in_w=45, out_b=46, out_w=47, vq0=48, vq1=49
  auto P = [&](int i) { return (const float*)d_in[2 + i]; };

  // ---- workspace carve ----
  size_t off = 0;
  auto carve = [&](size_t bytes) {
    void* p = (char*)d_ws + off;
    off += (bytes + 255) & ~(size_t)255;
    return p;
  };
  _Float16* bufA = (_Float16*)carve((size_t)B * T * D * 2);
  _Float16* bufB = (_Float16*)carve((size_t)B * T * D * 2);
  _Float16* bufC = (_Float16*)carve((size_t)B * T * D * 2);
  float* zres  = (float*)carve((size_t)B * 1024 * D * 4);
  float* quant = (float*)carve((size_t)B * 1024 * D * 4);
  float* xd    = (float*)carve((size_t)B * 256 * D * 4);
  _Float16* wh = (_Float16*)carve((size_t)8 * 1024 * 1024 * 2);
  float* cn0   = (float*)carve((size_t)4 * 1024 * 4);
  float* cn1   = (float*)carve((size_t)4 * 1024 * 4);
  int*   idx0  = (int*)carve((size_t)B * 256 * 4 * 4);
  int*   idx1  = (int*)carve((size_t)B * 1024 * 4 * 4);
  float* lacc  = (float*)carve(2 * 4);

  float* outY = (float*)d_out;
  const size_t yN = (size_t)B * T * C;
  float* outLoss = outY + yN;
  float* outI0 = outLoss + 1;
  float* outI1 = outI0 + (size_t)B * 256 * 4;

  // ---- pack weights/codebooks into fragment order ----
  size_t woff = 0;
  auto packConv = [&](const float* w, int Cout, int Cg, int k, int G) {
    _Float16* dst = wh + woff;
    const int Kt = Cg * k, KC = (Kt + 31) / 32;
    size_t n = (size_t)G * (Cout / G / 16) * KC * 512;
    woff += n;
    int ni = (int)n;
    pack_convB_kernel<<<dim3((ni + 255) / 256), dim3(256), 0, stream>>>(
        w, dst, Cout, Cg, k, G, ni);
    return (const _Float16*)dst;
  };
  auto packConvT = [&](const float* w, int Cin, int Cout, int s) {
    _Float16* dst = wh + woff;
    const int KC = (2 * Cin) / 32;
    size_t n = (size_t)s * (Cout / 16) * KC * 512;
    woff += n;
    int ni = (int)n;
    pack_convTB_kernel<<<dim3((ni + 255) / 256), dim3(256), 0, stream>>>(
        w, dst, Cin, Cout, s, ni);
    return (const _Float16*)dst;
  };
  auto packCB = [&](const float* cb) {
    _Float16* dst = wh + woff;
    size_t n = (size_t)4 * 64 * 4 * 512;
    woff += n;
    pack_cb_kernel<<<dim3(((int)n + 255) / 256), dim3(256), 0, stream>>>(
        cb, dst, (int)n);
    return (const _Float16*)dst;
  };

  const _Float16* Wenc_in  = packConv(P(45), D, C, 3, 1);
  const _Float16* We0r1c1  = packConv(P(27), D, D / GR, 3, GR);
  const _Float16* We0r1c2  = packConv(P(29), D, D / GR, 3, GR);
  const _Float16* We0r2c1  = packConv(P(31), D, D / GR, 3, GR);
  const _Float16* We0r2c2  = packConv(P(33), D, D / GR, 3, GR);
  const _Float16* We0dw    = packConv(P(25), D, D, 4, 1);
  const _Float16* We1r1c1  = packConv(P(37), D, D / GR, 3, GR);
  const _Float16* We1r1c2  = packConv(P(39), D, D / GR, 3, GR);
  const _Float16* We1r2c1  = packConv(P(41), D, D / GR, 3, GR);
  const _Float16* We1r2c2  = packConv(P(43), D, D / GR, 3, GR);
  const _Float16* We1dw    = packConv(P(35), D, D, 4, 1);
  const _Float16* Wenc_out = packConv(P(47), D, D, 3, 1);

  const _Float16* Wdec_in  = packConv(P(21), D, D, 3, 1);
  const _Float16* Wd0up    = packConvT(P(9), D, D, 2);
  const _Float16* Wd0r1c1  = packConv(P(1), D, D / GR, 3, GR);
  const _Float16* Wd0r1c2  = packConv(P(3), D, D / GR, 3, GR);
  const _Float16* Wd0r2c1  = packConv(P(5), D, D / GR, 3, GR);
  const _Float16* Wd0r2c2  = packConv(P(7), D, D / GR, 3, GR);
  const _Float16* Wd1up    = packConvT(P(19), D, D, 2);
  const _Float16* Wd1r1c1  = packConv(P(11), D, D / GR, 3, GR);
  const _Float16* Wd1r1c2  = packConv(P(13), D, D / GR, 3, GR);
  const _Float16* Wd1r2c1  = packConv(P(15), D, D / GR, 3, GR);
  const _Float16* Wd1r2c2  = packConv(P(17), D, D / GR, 3, GR);
  const _Float16* Wdec_out = packConv(P(23), C, D, 3, 1);

  const _Float16* CBT0 = packCB(P(48));
  const _Float16* CBT1 = packCB(P(49));
  cnorm_kernel<<<dim3(16), dim3(256), 0, stream>>>(P(48), cn0, 128, 4096);
  cnorm_kernel<<<dim3(16), dim3(256), 0, stream>>>(P(49), cn1, 128, 4096);

  // ---- typed conv launchers (block tile = 128 M x 16 N) ----
  auto convG = [&](const _Float16* in, const _Float16* w, const float* bias,
                   const _Float16* resid, _Float16* out, int Tl, int act) {
    dim3 grid(Tl / 128, D / 16, B);
    conv_wmma_kernel<0, 16, 3, true, true><<<grid, dim3(128), 0, stream>>>(
        in, w, bias, resid, out, Tl, Tl, D, D, 1, 0, act);
  };
  auto convD4s2 = [&](const _Float16* in, const _Float16* w, const float* bias,
                      _Float16* out, int Ti, int act) {
    dim3 grid((Ti / 2) / 128, D / 16, B);
    conv_wmma_kernel<0, 512, 4, true, true><<<grid, dim3(128), 0, stream>>>(
        in, w, bias, nullptr, out, Ti, Ti / 2, D, D, 2, 0, act);
  };
  auto convTup = [&](const _Float16* in, const _Float16* w, const float* bias,
                     _Float16* out, int Ti, int act) {
    const int s = 2, To = Ti * s, KC = (2 * D) / 32;
    for (int p = 0; p < s; ++p) {
      dim3 grid(Ti / 128, D / 16, B);
      conv_wmma_kernel<1, 512, 4, true, true><<<grid, dim3(128), 0, stream>>>(
          in, w + (size_t)p * (D / 16) * KC * 512, bias, nullptr, out,
          Ti, To, D, D, s, p, act);
    }
  };

  // ======================== Encoder ========================
  {  // input projection f32 -> f16 (Cin=80, k=3)
    dim3 grid(T / 128, D / 16, B);
    conv_wmma_kernel<0, 80, 3, false, true><<<grid, dim3(128), 0, stream>>>(
        x, Wenc_in, P(44), nullptr, bufA, T, T, C, D, 1, 0, ACT_SWISH);
  }
  convG(bufA, We0r1c1, P(26), nullptr, bufB, T, ACT_SWISH);
  convG(bufB, We0r1c2, P(28), bufA,    bufC, T, ACT_SWISH);
  convG(bufC, We0r2c1, P(30), nullptr, bufB, T, ACT_SWISH);
  convG(bufB, We0r2c2, P(32), bufC,    bufA, T, ACT_SWISH);
  convD4s2(bufA, We0dw, P(24), bufB, T, ACT_SWISH);
  convG(bufB, We1r1c1, P(36), nullptr, bufC, 2048, ACT_SWISH);
  convG(bufC, We1r1c2, P(38), bufB,    bufA, 2048, ACT_SWISH);
  convG(bufA, We1r2c1, P(40), nullptr, bufC, 2048, ACT_SWISH);
  convG(bufC, We1r2c2, P(42), bufA,    bufB, 2048, ACT_SWISH);
  convD4s2(bufB, We1dw, P(34), bufA, 2048, ACT_SWISH);
  {  // encoder out: f16 -> f32 z
    dim3 grid(1024 / 128, D / 16, B);
    conv_wmma_kernel<0, 512, 3, true, false><<<grid, dim3(128), 0, stream>>>(
        bufA, Wenc_out, P(46), nullptr, zres, 1024, 1024, D, D, 1, 0, ACT_NONE);
  }

  // ======================== Tree VQ ========================
  hipMemsetAsync(quant, 0, (size_t)B * 1024 * D * 4, stream);
  hipMemsetAsync(lacc, 0, 2 * 4, stream);
  {  // stage 0: rate 4
    const int rate = 4, rows = B * 256, n = rows * D;
    avgpool_kernel<<<dim3((n + 255) / 256), dim3(256), 0, stream>>>(
        zres, xd, 1024, D, rate, n);
    vq_argmin_kernel<<<dim3(rows / 16, 4), dim3(32), 0, stream>>>(
        xd, CBT0, cn0, idx0, outI0, rows, D);
    vq_loss_kernel<<<dim3((n + 255) / 256), dim3(256), 0, stream>>>(
        xd, P(48), idx0, lacc + 0, rows, 4, 1024, 128);
    const int nu = B * 1024 * D;
    vq_update_kernel<<<dim3((nu + 255) / 256), dim3(256), 0, stream>>>(
        zres, quant, P(48), idx0, 1024, D, rate, 4, 1024, 128, nu);
  }
  {  // stage 1: rate 1
    const int rows = B * 1024, n = rows * D;
    vq_argmin_kernel<<<dim3(rows / 16, 4), dim3(32), 0, stream>>>(
        zres, CBT1, cn1, idx1, outI1, rows, D);
    vq_loss_kernel<<<dim3((n + 255) / 256), dim3(256), 0, stream>>>(
        zres, P(49), idx1, lacc + 1, rows, 4, 1024, 128);
    vq_update_kernel<<<dim3((n + 255) / 256), dim3(256), 0, stream>>>(
        zres, quant, P(49), idx1, 1024, D, 1, 4, 1024, 128, n);
  }
  finalize_loss_kernel<<<dim3(1), dim3(1), 0, stream>>>(
      lacc, outLoss, (float)(B * 256 * D), (float)(B * 1024 * D));

  // ======================== Decoder ========================
  {  // decoder in: f32 quant -> f16
    dim3 grid(1024 / 128, D / 16, B);
    conv_wmma_kernel<0, 512, 3, false, true><<<grid, dim3(128), 0, stream>>>(
        quant, Wdec_in, P(20), nullptr, bufA, 1024, 1024, D, D, 1, 0, ACT_SWISH);
  }
  convTup(bufA, Wd0up, P(8), bufB, 1024, ACT_SWISH);
  convG(bufB, Wd0r1c1, P(0), nullptr, bufC, 2048, ACT_SWISH);
  convG(bufC, Wd0r1c2, P(2), bufB,    bufA, 2048, ACT_SWISH);
  convG(bufA, Wd0r2c1, P(4), nullptr, bufC, 2048, ACT_SWISH);
  convG(bufC, Wd0r2c2, P(6), bufA,    bufB, 2048, ACT_SWISH);
  convTup(bufB, Wd1up, P(18), bufA, 2048, ACT_SWISH);
  convG(bufA, Wd1r1c1, P(10), nullptr, bufC, 4096, ACT_SWISH);
  convG(bufC, Wd1r1c2, P(12), bufA,    bufB, 4096, ACT_SWISH);
  convG(bufB, Wd1r2c1, P(14), nullptr, bufC, 4096, ACT_SWISH);
  convG(bufC, Wd1r2c2, P(16), bufB,    bufA, 4096, ACT_SWISH);
  {  // output projection -> y [B,T,C=80] f32 directly into d_out
    dim3 grid(T / 128, C / 16, B);
    conv_wmma_kernel<0, 512, 3, true, false><<<grid, dim3(128), 0, stream>>>(
        bufA, Wdec_out, P(22), nullptr, outY, T, T, D, C, 1, 0, ACT_NONE);
  }
}
